// nGPT_41635412967754
// MI455X (gfx1250) — compile-verified, matrix operand order checked by reading, not run on hardware
//
#include <hip/hip_runtime.h>
#include <hip/hip_bf16.h>
#include <math.h>

typedef _Float16 v16h __attribute__((ext_vector_type(16)));
typedef _Float16 h8   __attribute__((ext_vector_type(8)));
typedef float    v8f  __attribute__((ext_vector_type(8)));

#if defined(__has_builtin)
#if __has_builtin(__builtin_amdgcn_sched_barrier)
#define SCHED_FENCE() __builtin_amdgcn_sched_barrier(0)
#endif
#endif
#ifndef SCHED_FENCE
#define SCHED_FENCE() ((void)0)
#endif

#define NL   4
#define ND   1024
#define NH   16
#define NDH  64
#define NDI  1024
#define NDFF 2730
#define NDFFP 2736   /* padded leading dim for FF activations (16B-aligned rows) */
#define NV   32000
#define NB   2
#define NS   512
#define NT   (NB*NS)   /* 1024 tokens */

// ---- reference-exact l2norm scale: t / max(n / clamp(n,1-eps,1+eps), 1e-10) ----
__device__ __forceinline__ float l2scale_from_norm(float n) {
  float tgt = fminf(fmaxf(n, 1.0f - 1e-5f), 1.0f + 1e-5f);
  return 1.0f / fmaxf(n / tgt, 1e-10f);
}

__device__ __forceinline__ float block_reduce_sum(float v, float* red) {
  int tid = threadIdx.x;
  red[tid] = v;
  __syncthreads();
  for (int s = blockDim.x >> 1; s > 0; s >>= 1) {
    if (tid < s) red[tid] += red[tid + s];
    __syncthreads();
  }
  float r = red[0];
  __syncthreads();
  return r;
}

// ---------- weight-norm prologue ----------
__global__ __launch_bounds__(256) void rownorm_kernel(const float* __restrict__ W,
                                                      float* __restrict__ rs, int K) {
  __shared__ float red[256];
  int row = blockIdx.x;
  const float* p = W + (size_t)row * K;
  float s = 0.0f;
  for (int j = threadIdx.x; j < K; j += 256) { float w = p[j]; s += w * w; }
  float tot = block_reduce_sum(s, red);
  if (threadIdx.x == 0) rs[row] = l2scale_from_norm(sqrtf(tot));
}

// per-column reciprocal l2norm scale (W is R x C row-major, norm over R)
__global__ __launch_bounds__(256) void colnorm_kernel(const float* __restrict__ W,
                                                      float* __restrict__ rs, int R, int C) {
  int c = blockIdx.x * 256 + threadIdx.x;
  if (c >= C) return;
  float s = 0.0f;
  for (int r = 0; r < R; ++r) { float w = W[(size_t)r * C + c]; s += w * w; }
  rs[c] = l2scale_from_norm(sqrtf(s));
}

// ---------- embedding gather + row l2norm ----------
__global__ __launch_bounds__(256) void embed_kernel(const int* __restrict__ ids,
                                                    const float* __restrict__ W_emb,
                                                    float* __restrict__ x) {
  __shared__ float red[256];
  int tok = blockIdx.x;
  int id = ids[tok];
  const float* row = W_emb + (size_t)id * ND;
  float vals[4]; float s = 0.0f;
  #pragma unroll
  for (int i = 0; i < 4; ++i) { vals[i] = row[threadIdx.x + i * 256]; s += vals[i] * vals[i]; }
  float sc = l2scale_from_norm(sqrtf(block_reduce_sum(s, red)));
  #pragma unroll
  for (int i = 0; i < 4; ++i) x[(size_t)tok * ND + threadIdx.x + i * 256] = vals[i] * sc;
}

// ---------- WMMA GEMM: C[M,N] = A[M,K] * (B[N,K] .* norm-scale)^T ----------
// fp32 in HBM, f16 in LDS/WMMA, f32 accumulate. 128x128 block tile, K-step 32.
// Software-pipelined: double-buffered LDS; staging loads pinned before the
// WMMA block (sched_barrier) and global_prefetch_b8 issued two K-steps ahead
// so late-scheduled loads hit L2/WGP$ instead of paying HBM latency.
// NG : N not multiple of 128 (clamped rows, zero folded into scale)
// KG : K not multiple of 32 (clamped scalar loads, zero multiplier at cvt)
// SOK: B scale indexed by K (column-normalized weights) vs by N (row-normalized)
template <bool NG, bool KG, bool SOK>
__global__ __launch_bounds__(256) __attribute__((amdgpu_waves_per_eu(2)))
void gemm_wmma_kernel(
    const float* __restrict__ A, int lda,
    const float* __restrict__ B, int ldb,
    const float* __restrict__ bscale,
    const float* __restrict__ oscale, float oscale_mul,
    float* __restrict__ C, int ldc,
    int N, int K)
{
  __shared__ _Float16 As[2][128 * 40];   // stride 40 halves: conflict-free b128 reads
  __shared__ _Float16 Bs[2][128 * 40];
  int tid = threadIdx.x;
  int nBase = blockIdx.x * 128;
  int mBase = blockIdx.y * 128;
  int wave = tid >> 5, lane = tid & 31;
  int wm = wave & 1, wn = wave >> 1;           // 2 (M) x 4 (N) waves
  int lrow = lane & 15, khalf = lane >> 4;

  v8f acc[4][2];
  #pragma unroll
  for (int mi = 0; mi < 4; ++mi)
    #pragma unroll
    for (int ni = 0; ni < 2; ++ni)
      #pragma unroll
      for (int e = 0; e < 8; ++e) acc[mi][ni][e] = 0.0f;

  // per-thread staging coordinates (chunk of 8 contiguous K per row)
  int rowA = tid >> 2;
  int kcA  = (tid & 3) * 8;

  // in-flight staging registers (raw fp32; scaling/cvt deferred to store phase)
  float aR[2][8], bR[2][8], sV[2][8], sS[2];

  // ---- load phase: issue global loads only; weights (B, HBM stream) first ----
  auto load_regs = [&](int k0) {
    #pragma unroll
    for (int it = 0; it < 2; ++it) {
      int row = rowA + it * 64;
      int gn = nBase + row;
      int gnc = NG ? ((gn < N) ? gn : (N - 1)) : gn;
      const float* rowpB = B + (size_t)gnc * ldb;
      if (!KG) {
        const float* bp = rowpB + (k0 + kcA);
        float4 b0 = *(const float4*)bp;
        float4 b1 = *(const float4*)(bp + 4);
        bR[it][0]=b0.x; bR[it][1]=b0.y; bR[it][2]=b0.z; bR[it][3]=b0.w;
        bR[it][4]=b1.x; bR[it][5]=b1.y; bR[it][6]=b1.z; bR[it][7]=b1.w;
        if (SOK) {
          float4 s0 = *(const float4*)(bscale + k0 + kcA);
          float4 s1 = *(const float4*)(bscale + k0 + kcA + 4);
          sV[it][0]=s0.x; sV[it][1]=s0.y; sV[it][2]=s0.z; sV[it][3]=s0.w;
          sV[it][4]=s1.x; sV[it][5]=s1.y; sV[it][6]=s1.z; sV[it][7]=s1.w;
        } else {
          sS[it] = bscale[gnc];
        }
      } else {
        #pragma unroll
        for (int j = 0; j < 8; ++j) {
          int kg = k0 + kcA + j;
          int kcl = (kg < K) ? kg : (K - 1);
          bR[it][j] = rowpB[kcl];
          if (SOK) sV[it][j] = bscale[kcl];
        }
        if (!SOK) sS[it] = bscale[gnc];
      }
    }
    #pragma unroll
    for (int it = 0; it < 2; ++it) {
      int row = rowA + it * 64;
      const float* rowpA = A + (size_t)(mBase + row) * lda;
      if (!KG) {
        const float* ap = rowpA + (k0 + kcA);
        float4 a0 = *(const float4*)ap;
        float4 a1 = *(const float4*)(ap + 4);
        aR[it][0]=a0.x; aR[it][1]=a0.y; aR[it][2]=a0.z; aR[it][3]=a0.w;
        aR[it][4]=a1.x; aR[it][5]=a1.y; aR[it][6]=a1.z; aR[it][7]=a1.w;
      } else {
        #pragma unroll
        for (int j = 0; j < 8; ++j) {
          int kg = k0 + kcA + j;
          aR[it][j] = rowpA[(kg < K) ? kg : (K - 1)];
        }
      }
    }
    // prefetch tile i+2 (uniform guard keeps addresses in-bounds; lowers to
    // global_prefetch_b8: no VGPR result, no LOADcnt, cannot be "waited on")
    int kpf = k0 + 64;
    if (kpf + 32 <= K) {
      #pragma unroll
      for (int it = 0; it < 2; ++it) {
        int row = rowA + it * 64;
        int gn = nBase + row;
        int gnc = NG ? ((gn < N) ? gn : (N - 1)) : gn;
        __builtin_prefetch(B + (size_t)gnc * ldb + kpf + kcA, 0, 2);
        __builtin_prefetch(A + (size_t)(mBase + row) * lda + kpf + kcA, 0, 2);
      }
    }
  };

  // ---- store phase: convert (folding scales/masks) and write LDS buffer ----
  // B first (matches load issue order so loadcnt waits retire incrementally).
  auto store_regs = [&](int buf, int k0) {
    #pragma unroll
    for (int it = 0; it < 2; ++it) {
      int row = rowA + it * 64;
      int gn = nBase + row;
      h8 hv;
      if (SOK) {
        #pragma unroll
        for (int j = 0; j < 8; ++j) {
          float m = (!KG || (k0 + kcA + j < K)) ? 1.0f : 0.0f;
          hv[j] = (_Float16)(bR[it][j] * sV[it][j] * m);
        }
      } else {
        float rs = sS[it];
        if (NG) rs = (gn < N) ? rs : 0.0f;   // zero out-of-range rows, branch-free
        #pragma unroll
        for (int j = 0; j < 8; ++j) {
          float m = (!KG || (k0 + kcA + j < K)) ? 1.0f : 0.0f;
          hv[j] = (_Float16)(bR[it][j] * rs * m);
        }
      }
      *(h8*)&Bs[buf][row * 40 + kcA] = hv;
    }
    #pragma unroll
    for (int it = 0; it < 2; ++it) {
      int row = rowA + it * 64;
      h8 hv;
      #pragma unroll
      for (int j = 0; j < 8; ++j) {
        float m = (!KG || (k0 + kcA + j < K)) ? 1.0f : 0.0f;
        hv[j] = (_Float16)(aR[it][j] * m);
      }
      *(h8*)&As[buf][row * 40 + kcA] = hv;
    }
  };

  // ---- compute phase: fragments + 8 WMMAs per wave ----
  auto compute = [&](int buf) {
    v16h bfrag[2];
    #pragma unroll
    for (int ni = 0; ni < 2; ++ni) {
      const _Float16* p = &Bs[buf][(wn * 32 + ni * 16 + lrow) * 40 + khalf * 8];
      h8 lo = *(const h8*)p;
      h8 hi = *(const h8*)(p + 16);
      #pragma unroll
      for (int e = 0; e < 8; ++e) { bfrag[ni][e] = lo[e]; bfrag[ni][e + 8] = hi[e]; }
    }
    #pragma unroll
    for (int mi = 0; mi < 4; ++mi) {
      const _Float16* p = &As[buf][(wm * 64 + mi * 16 + lrow) * 40 + khalf * 8];
      h8 lo = *(const h8*)p;
      h8 hi = *(const h8*)(p + 16);
      v16h afrag;
      #pragma unroll
      for (int e = 0; e < 8; ++e) { afrag[e] = lo[e]; afrag[e + 8] = hi[e]; }
      acc[mi][0] = __builtin_amdgcn_wmma_f32_16x16x32_f16(
          false, afrag, false, bfrag[0], (short)0, acc[mi][0], false, false);
      acc[mi][1] = __builtin_amdgcn_wmma_f32_16x16x32_f16(
          false, afrag, false, bfrag[1], (short)0, acc[mi][1], false, false);
    }
  };

  // ---- pipelined main loop: one barrier per K-step ----
  load_regs(0);
  store_regs(0, 0);
  __syncthreads();
  int cur = 0;
  for (int k0 = 32; k0 < K; k0 += 32) {
    load_regs(k0);        // global loads issued...
    SCHED_FENCE();        // ...and pinned before the WMMA block
    compute(cur);         // WMMAs run with the loads still in flight
    SCHED_FENCE();        // keep loadcnt waits (cvt/store) after the WMMAs
    store_regs(cur ^ 1, k0);
    __syncthreads();
    cur ^= 1;
  }
  compute(cur);

  // ---- write C (optional per-N output scale, e.g. logit_scale * sqrt(D)) ----
  #pragma unroll
  for (int mi = 0; mi < 4; ++mi) {
    #pragma unroll
    for (int ni = 0; ni < 2; ++ni) {
      int nc = nBase + wn * 32 + ni * 16 + lrow;
      if (!NG || nc < N) {
        float os = oscale ? (oscale[nc] * oscale_mul) : 1.0f;
        int m0 = mBase + wm * 64 + mi * 16 + khalf * 8;
        #pragma unroll
        for (int r = 0; r < 8; ++r)
          C[(size_t)(m0 + r) * ldc + nc] = acc[mi][ni][r] * os;
      }
    }
  }
}

// ---------- value residual mix ----------
__global__ __launch_bounds__(256) void vmix_kernel(float* __restrict__ v,
                                                   float* __restrict__ v0, int mode) {
  int i = blockIdx.x * 256 + threadIdx.x;
  if (i >= NT * NDI) return;
  if (mode == 0) v0[i] = v[i];
  else v[i] = 0.5f * (v[i] + v0[i]);
}

// ---------- per-(token,head) l2norm of q,k + beta + qk scale ----------
__global__ __launch_bounds__(64) void headnorm_kernel(
    float* __restrict__ q, float* __restrict__ k, float* __restrict__ beta,
    const float* __restrict__ wbeta_l, const float* __restrict__ qk_scale_l,
    const float* __restrict__ beta_scale_p)
{
  __shared__ float red[64];
  int tok = blockIdx.x >> 4;
  int h = blockIdx.x & 15;
  int d = threadIdx.x;
  size_t idx = (size_t)tok * NDI + h * NDH + d;
  float qv = q[idx], kv = k[idx], wb = wbeta_l[d];
  float nq = sqrtf(block_reduce_sum(qv * qv, red));
  float nk = sqrtf(block_reduce_sum(kv * kv, red));
  float nw = sqrtf(block_reduce_sum(wb * wb, red));
  float qn = qv * l2scale_from_norm(nq);
  float kn = kv * l2scale_from_norm(nk);
  float wn = wb * l2scale_from_norm(nw);
  float dot = block_reduce_sum(kn * wn, red);
  q[idx] = qn * qk_scale_l[h * NDH + d] * (float)ND;   // Scale forward = param * D
  k[idx] = kn;
  if (d == 0) {
    float z = dot * beta_scale_p[0] * 0.1f;
    beta[tok * NH + h] = 1.0f / (1.0f + expf(-z));
  }
}

// ---------- delta-rule scan: independent state columns, 4 cols / wave ----------
__global__ __launch_bounds__(32) void delta_kernel(
    const float* __restrict__ q, const float* __restrict__ k,
    const float* __restrict__ v, const float* __restrict__ beta,
    float* __restrict__ o)
{
  int blk = blockIdx.x;
  int vblk = blk & 15;
  int h = (blk >> 4) & 15;
  int b = blk >> 8;
  int lane = threadIdx.x;
  int vi = lane & 3, ki = lane >> 2;
  int vcol = vblk * 4 + vi;
  int base_h = h * NDH;

  float S[8];
  #pragma unroll
  for (int j = 0; j < 8; ++j) S[j] = 0.0f;

  for (int t = 0; t < NS; ++t) {
    int tok = b * NS + t;
    const float* kp = k + (size_t)tok * NDI + base_h + ki * 8;
    const float* qp = q + (size_t)tok * NDI + base_h + ki * 8;
    float4 ka = *(const float4*)kp;  float4 kb = *(const float4*)(kp + 4);
    float4 qa = *(const float4*)qp;  float4 qb = *(const float4*)(qp + 4);
    float kk[8] = {ka.x, ka.y, ka.z, ka.w, kb.x, kb.y, kb.z, kb.w};
    float qq[8] = {qa.x, qa.y, qa.z, qa.w, qb.x, qb.y, qb.z, qb.w};
    float vt = v[(size_t)tok * NDI + base_h + vcol];
    float bt = beta[tok * NH + h];

    float ks = 0.0f;
    #pragma unroll
    for (int j = 0; j < 8; ++j) ks += kk[j] * S[j];
    ks += __shfl_xor(ks, 4);
    ks += __shfl_xor(ks, 8);
    ks += __shfl_xor(ks, 16);

    float dlt = bt * (vt - ks);
    float ov = 0.0f;
    #pragma unroll
    for (int j = 0; j < 8; ++j) { S[j] += kk[j] * dlt; ov += qq[j] * S[j]; }
    ov += __shfl_xor(ov, 4);
    ov += __shfl_xor(ov, 8);
    ov += __shfl_xor(ov, 16);
    if (ki == 0) o[(size_t)tok * NDI + base_h + vcol] = ov;
  }
}

// ---------- residual: x = l2norm(x + a*(l2norm(h) - x)) ----------
__global__ __launch_bounds__(256) void resnorm_kernel(
    float* __restrict__ x, const float* __restrict__ hin,
    const float* __restrict__ alpha, float alpha_fs)
{
  __shared__ float red[256];
  int tok = blockIdx.x;
  float hv[4], xv[4];
  float s = 0.0f;
  #pragma unroll
  for (int i = 0; i < 4; ++i) {
    int d = threadIdx.x + i * 256;
    hv[i] = hin[(size_t)tok * ND + d];
    s += hv[i] * hv[i];
  }
  float hs = l2scale_from_norm(sqrtf(block_reduce_sum(s, red)));
  s = 0.0f;
  #pragma unroll
  for (int i = 0; i < 4; ++i) {
    int d = threadIdx.x + i * 256;
    float xo = x[(size_t)tok * ND + d];
    float y = xo + alpha[d] * alpha_fs * (hv[i] * hs - xo);
    xv[i] = y;
    s += y * y;
  }
  float xs = l2scale_from_norm(sqrtf(block_reduce_sum(s, red)));
  #pragma unroll
  for (int i = 0; i < 4; ++i)
    x[(size_t)tok * ND + threadIdx.x + i * 256] = xv[i] * xs;
}

// ---------- SiLU gating: f = silu(gate*gs*sqrt(D)) * (hidden*hs), in-place over hidden ----------
__global__ __launch_bounds__(256) void gate_kernel(
    const float* __restrict__ hid, const float* __restrict__ gat,
    const float* __restrict__ ffh_s, const float* __restrict__ ffg_s,
    float* __restrict__ f)
{
  int col = blockIdx.x * 256 + threadIdx.x;
  if (col >= NDFF) return;
  int row = blockIdx.y;
  size_t i = (size_t)row * NDFFP + col;
  float g = gat[i] * ffg_s[col] * 32.0f;       // sqrt(D) = 32
  float hv = hid[i] * ffh_s[col];
  float sg = g / (1.0f + expf(-g));            // silu
  f[i] = sg * hv;
}

extern "C" void kernel_launch(void* const* d_in, const int* in_sizes, int n_in,
                              void* d_out, int out_size, void* d_ws, size_t ws_size,
                              hipStream_t stream) {
  (void)in_sizes; (void)n_in; (void)out_size; (void)ws_size;
  const int*   ids         = (const int*)  d_in[0];
  /* d_in[1] offsets unused (equal-length rows) */
  const float* W_emb       = (const float*)d_in[2];
  const float* Wq          = (const float*)d_in[3];
  const float* Wk          = (const float*)d_in[4];
  const float* Wv          = (const float*)d_in[5];
  const float* Wbeta       = (const float*)d_in[6];
  const float* Wo          = (const float*)d_in[7];
  const float* Wffh        = (const float*)d_in[8];
  const float* Wffg        = (const float*)d_in[9];
  const float* Wffo        = (const float*)d_in[10];
  const float* qk_scale    = (const float*)d_in[11];
  const float* beta_scale  = (const float*)d_in[12];
  const float* attn_alpha  = (const float*)d_in[13];
  const float* ff_alpha    = (const float*)d_in[14];
  const float* ffh_scale   = (const float*)d_in[15];
  const float* ffg_scale   = (const float*)d_in[16];
  const float* W_logits    = (const float*)d_in[17];
  const float* logit_scale = (const float*)d_in[18];
  float* out = (float*)d_out;

  float* ws = (float*)d_ws;
  size_t off = 0;
  auto carve = [&](size_t n) -> float* {
    float* p = ws + off;
    off += (n + 63) & ~(size_t)63;
    return p;
  };
  float* rWq   = carve((size_t)NL * NDI);
  float* rWk   = carve((size_t)NL * NDI);
  float* rWv   = carve((size_t)NL * NDI);
  float* rWo   = carve((size_t)NL * NDI);    // per-K (column norms of Wo)
  float* rWffh = carve((size_t)NL * NDFF);
  float* rWffg = carve((size_t)NL * NDFF);
  float* rWffo = carve((size_t)NL * NDFF);   // per-K (column norms of Wffo)
  float* rWlog = carve((size_t)NV);
  float* x    = carve((size_t)NT * ND);
  float* q    = carve((size_t)NT * NDI);
  float* k    = carve((size_t)NT * NDI);
  float* v    = carve((size_t)NT * NDI);
  float* v0   = carve((size_t)NT * NDI);
  float* beta = carve((size_t)NT * NH);
  float* o    = carve((size_t)NT * NDI);
  float* t1   = carve((size_t)NT * ND);
  float* hid  = carve((size_t)NT * NDFFP);   // padded ld so rows are 16B aligned
  float* gat  = carve((size_t)NT * NDFFP);
  float* f    = hid;   // in-place gating

  const float ALPHA_FS = 8.0f;  // (1/L)/D^-0.5 = 0.25*32

  // ---- weight norm prologue ----
  rownorm_kernel<<<NL * NDI, 256, 0, stream>>>(Wq, rWq, ND);
  rownorm_kernel<<<NL * NDI, 256, 0, stream>>>(Wk, rWk, ND);
  rownorm_kernel<<<NL * NDI, 256, 0, stream>>>(Wv, rWv, ND);
  rownorm_kernel<<<NL * NDFF, 256, 0, stream>>>(Wffh, rWffh, ND);
  rownorm_kernel<<<NL * NDFF, 256, 0, stream>>>(Wffg, rWffg, ND);
  rownorm_kernel<<<NV, 256, 0, stream>>>(W_logits, rWlog, ND);
  for (int l = 0; l < NL; ++l) {
    colnorm_kernel<<<(NDI + 255) / 256, 256, 0, stream>>>(
        Wo + (size_t)l * ND * NDI, rWo + l * NDI, ND, NDI);
    colnorm_kernel<<<(NDFF + 255) / 256, 256, 0, stream>>>(
        Wffo + (size_t)l * ND * NDFF, rWffo + l * NDFF, ND, NDFF);
  }

  // ---- embedding ----
  embed_kernel<<<NT, 256, 0, stream>>>(ids, W_emb, x);

  dim3 g1024(NDI / 128, NT / 128);            // 8 x 8
  dim3 gFF((NDFF + 127) / 128, NT / 128);     // 22 x 8
  dim3 gV(NV / 128, NT / 128);                // 250 x 8

  for (int l = 0; l < NL; ++l) {
    const float* Wq_l   = Wq   + (size_t)l * NDI * ND;
    const float* Wk_l   = Wk   + (size_t)l * NDI * ND;
    const float* Wv_l   = Wv   + (size_t)l * NDI * ND;
    const float* Wo_l   = Wo   + (size_t)l * ND * NDI;
    const float* Wffh_l = Wffh + (size_t)l * NDFF * ND;
    const float* Wffg_l = Wffg + (size_t)l * NDFF * ND;
    const float* Wffo_l = Wffo + (size_t)l * ND * NDFF;

    // q,k,v projections (row-normalized weights): fast guard-free path
    gemm_wmma_kernel<false,false,false><<<g1024, 256, 0, stream>>>(
        x, ND, Wq_l, ND, rWq + l * NDI, nullptr, 1.0f, q, NDI, NDI, ND);
    gemm_wmma_kernel<false,false,false><<<g1024, 256, 0, stream>>>(
        x, ND, Wk_l, ND, rWk + l * NDI, nullptr, 1.0f, k, NDI, NDI, ND);
    gemm_wmma_kernel<false,false,false><<<g1024, 256, 0, stream>>>(
        x, ND, Wv_l, ND, rWv + l * NDI, nullptr, 1.0f, v, NDI, NDI, ND);
    vmix_kernel<<<(NT * NDI) / 256, 256, 0, stream>>>(v, v0, (l == 0) ? 0 : 1);
    headnorm_kernel<<<NT * NH, 64, 0, stream>>>(q, k, beta, Wbeta + l * NDH,
                                                qk_scale + l * NDI, beta_scale + l);
    delta_kernel<<<NB * NH * 16, 32, 0, stream>>>(q, k, v, beta, o);
    // output projection (column-normalized Wo: scale indexed by K)
    gemm_wmma_kernel<false,false,true><<<g1024, 256, 0, stream>>>(
        o, NDI, Wo_l, NDI, rWo + l * NDI, nullptr, 1.0f, t1, ND, ND, NDI);
    resnorm_kernel<<<NT, 256, 0, stream>>>(x, t1, attn_alpha + l * ND, ALPHA_FS);
    // feedforward: N=2730 needs row guard (clamped, branch-free)
    gemm_wmma_kernel<true,false,false><<<gFF, 256, 0, stream>>>(
        x, ND, Wffh_l, ND, rWffh + l * NDFF, nullptr, 1.0f, hid, NDFFP, NDFF, ND);
    gemm_wmma_kernel<true,false,false><<<gFF, 256, 0, stream>>>(
        x, ND, Wffg_l, ND, rWffg + l * NDFF, nullptr, 1.0f, gat, NDFFP, NDFF, ND);
    gate_kernel<<<dim3((NDFF + 255) / 256, NT), 256, 0, stream>>>(
        hid, gat, ffh_scale + l * NDFF, ffg_scale + l * NDFF, f);
    // ffo: K=2730 needs K guard (clamped scalar staging, per-K scale)
    gemm_wmma_kernel<false,true,true><<<g1024, 256, 0, stream>>>(
        f, NDFFP, Wffo_l, NDFF, rWffo + l * NDFF, nullptr, 1.0f, t1, ND, ND, NDFF);
    resnorm_kernel<<<NT, 256, 0, stream>>>(x, t1, ff_alpha + l * ND, ALPHA_FS);
  }

  // ---- logits straight into d_out, logit_scale*sqrt(D) fused as output scale ----
  gemm_wmma_kernel<false,false,false><<<gV, 256, 0, stream>>>(
      x, ND, W_logits, ND, rWlog, logit_scale, 32.0f, out, NV, NV, ND);
}
